// DecoderUnit_54494545051892
// MI455X (gfx1250) — compile-verified, matrix operand order checked by reading, not run on hardware
//
#include <hip/hip_runtime.h>
#include <hip/hip_bf16.h>

// ---------------- problem constants (match reference) ----------------
constexpr int NB   = 16;
constexpr int NT   = 128;
constexpr int NV   = 25;
constexpr int NCIN = 64;
constexpr int NCM  = 256;
constexpr int NH   = 8;
constexpr int ND   = 32;
constexpr int NHD  = NH * ND;        // 256
constexpr int NK   = 3;
constexpr int NCOUT= 256;
constexpr int ROWS = NB * NT * NV;   // 51200
constexpr float EPSLN = 1e-6f;

typedef __attribute__((ext_vector_type(16))) _Float16 v16h;
typedef __attribute__((ext_vector_type(8)))  _Float16 v8h;
typedef __attribute__((ext_vector_type(8)))  float    v8f;

// -------- CDNA5 async copy (global -> LDS), ASYNCcnt protocol --------
__device__ __forceinline__ uint32_t lds_off_u32(const void* p) {
  // generic LDS pointer: low 32 bits are the wave-relative LDS byte offset
  return (uint32_t)(uintptr_t)p;
}
__device__ __forceinline__ void async_ld_b128(uint32_t lds_byte_off, const void* g) {
  asm volatile("global_load_async_to_lds_b128 %0, %1, off"
               :: "v"(lds_byte_off), "v"(g) : "memory");
}
__device__ __forceinline__ void wait_async0() {
  asm volatile("s_wait_asynccnt 0x0" ::: "memory");
}

// WMMA 16x16x32 f16 fragment K-offset for a lane (A and B share the pattern
// when B is stored transposed [N,K]).
__device__ __forceinline__ int wmma_kbase(int lane) { return (lane & 16) ? 8 : 0; }

__device__ __forceinline__ v16h frag_from_lds(const _Float16* base /* row start */, int kbase) {
  v8h lo = *(const v8h*)(base + kbase);
  v8h hi = *(const v8h*)(base + 16 + kbase);
  v16h f;
#pragma unroll
  for (int j = 0; j < 8; ++j) { f[j] = lo[j]; f[8 + j] = hi[j]; }
  return f;
}

// ---------------------------------------------------------------
// f16 GEMM: C[M,N] = Ah[M,K] * Bt[N,K]^T (+bias). Block tile 128x64,
// 8 waves (4x2), each wave 32x32 = 2x2 WMMA tiles. Double-buffered
// LDS tiles filled by async DMA so DMA(k+1) overlaps WMMA(k).
// Requires M%128==0, N%64==0, K%32==0.
// ---------------------------------------------------------------
template <bool HASBIAS>
__global__ void __launch_bounds__(256)
gemm_wmma_f16(const _Float16* __restrict__ Ah,
              const _Float16* __restrict__ Bt,
              const float* __restrict__ bias,
              float* __restrict__ C,
              int M, int N, int Kd) {
  __shared__ __align__(16) _Float16 Alds[2][128 * 32];
  __shared__ __align__(16) _Float16 Blds[2][64 * 32];
  const int tid  = threadIdx.x;
  const int lane = tid & 31, wave = tid >> 5;
  const int waveM = wave >> 1, waveN = wave & 1;
  const int nBlocksN = N >> 6;
  const int bm = blockIdx.x / nBlocksN;
  const int bn = blockIdx.x % nBlocksN;
  const int row0 = bm * 128, col0 = bn * 64;

  // staging assignment: A 2 thr/row (32B each), B 4 thr/row (16B each)
  const int ar = tid >> 1, apart = tid & 1;
  const int br = tid >> 2, bpart = tid & 3;
  const int kbase = wmma_kbase(lane);

  auto stage = [&](int buf, int k0) {
    const _Float16* ga = Ah + (long)(row0 + ar) * Kd + k0 + apart * 16;
    const uint32_t la = lds_off_u32(&Alds[buf][ar * 32 + apart * 16]);
    async_ld_b128(la, ga);
    async_ld_b128(la + 16u, (const char*)ga + 16);
    async_ld_b128(lds_off_u32(&Blds[buf][br * 32 + bpart * 8]),
                  Bt + (long)(col0 + br) * Kd + k0 + bpart * 8);
  };

  v8f acc[2][2] = {};
  stage(0, 0);                       // prologue: tile 0 in flight
  int cur = 0;
  for (int k0 = 0; k0 < Kd; k0 += 32) {
    wait_async0();                   // tile k landed (in-order async completion)
    __syncthreads();                 // visible to all waves; prev reads of cur^1 done
    if (k0 + 32 < Kd) stage(cur ^ 1, k0 + 32);   // DMA next tile during compute

    v16h afr[2], bfr[2];
#pragma unroll
    for (int s = 0; s < 2; ++s) {
      const int m = waveM * 32 + s * 16 + (lane & 15);
      afr[s] = frag_from_lds(&Alds[cur][m * 32], kbase);
      const int n = waveN * 32 + s * 16 + (lane & 15);
      bfr[s] = frag_from_lds(&Blds[cur][n * 32], kbase);
    }
#pragma unroll
    for (int sm = 0; sm < 2; ++sm)
#pragma unroll
      for (int sn = 0; sn < 2; ++sn)
        acc[sm][sn] = __builtin_amdgcn_wmma_f32_16x16x32_f16(
            false, afr[sm], false, bfr[sn], (short)0, acc[sm][sn], false, false);
    cur ^= 1;
  }

  const int rbase = (lane & 16) ? 8 : 0;
#pragma unroll
  for (int sm = 0; sm < 2; ++sm)
#pragma unroll
    for (int sn = 0; sn < 2; ++sn) {
      const int col = col0 + waveN * 32 + sn * 16 + (lane & 15);
      const float bv = HASBIAS ? bias[col] : 0.0f;
#pragma unroll
      for (int r = 0; r < 8; ++r) {
        const int row = row0 + waveM * 32 + sm * 16 + rbase + r;
        C[(long)row * N + col] = acc[sm][sn][r] + bv;
      }
    }
}

// ---------------------------------------------------------------
// Fused temporal attention for one (b,h,v): 128 queries x 128 keys,
// head dim 32. LDS: Qh/Kh (t-major), Vt (d-major, transposed), S f32,
// P f16. All WMMA fragments are contiguous ds_load_b128 pairs.
// ---------------------------------------------------------------
__global__ void __launch_bounds__(256)
attn_wmma(const float* __restrict__ Qb, const float* __restrict__ Kb,
          const float* __restrict__ Vb, float* __restrict__ Ob, int causal) {
  extern __shared__ __align__(16) char smem[];
  _Float16* Qh = (_Float16*)smem;                    // [128][32]
  _Float16* Kh = Qh + NT * ND;                       // [128][32]
  _Float16* Vt = Kh + NT * ND;                       // [32][128] (transposed)
  float*    S  = (float*)(smem + 3 * NT * ND * 2);   // [128][128] f32
  _Float16* P  = (_Float16*)(S + NT * NT);           // [128][128] f16

  const int blk = blockIdx.x;
  const int b = blk / (NH * NV);
  const int rem = blk % (NH * NV);
  const int h = rem / NV;
  const int v = rem % NV;
  const int tid = threadIdx.x;
  const int lane = tid & 31, wave = tid >> 5;
  const int kbase = wmma_kbase(lane);

  for (int e = tid; e < NT * ND; e += 256) {
    const int t = e >> 5, d = e & 31;
    const long base = ((long)(b * NT + t) * NV + v) * NHD + h * ND + d;
    Qh[e] = (_Float16)Qb[base];
    Kh[e] = (_Float16)Kb[base];
    Vt[d * NT + t] = (_Float16)Vb[base];
  }
  __syncthreads();

  // scores: 64 16x16 tiles, 8 per wave, K=32 in one WMMA
  const float scale = 0.17677669529663687f;
#pragma unroll 1
  for (int i = 0; i < 8; ++i) {
    const int tile = wave * 8 + i;
    const int ti = tile >> 3, tj = tile & 7;
    const int m = ti * 16 + (lane & 15);
    const int n = tj * 16 + (lane & 15);
    v16h af = frag_from_lds(&Qh[m * ND], kbase);
    v16h bf = frag_from_lds(&Kh[n * ND], kbase);
    v8f c = {};
    c = __builtin_amdgcn_wmma_f32_16x16x32_f16(false, af, false, bf,
                                               (short)0, c, false, false);
    const int rbase = (lane & 16) ? 8 : 0;
#pragma unroll
    for (int r = 0; r < 8; ++r) {
      const int row = ti * 16 + rbase + r;
      const int col = tj * 16 + (lane & 15);
      float sv = c[r] * scale;
      if (causal && col > row) sv = -1e9f;
      S[row * NT + col] = sv;
    }
  }
  __syncthreads();

  if (tid < NT) {
    const int r = tid;
    float mx = -3.4e38f;
    for (int cc = 0; cc < NT; ++cc) mx = fmaxf(mx, S[r * NT + cc]);
    float sum = 0.f;
    for (int cc = 0; cc < NT; ++cc) {
      const float e = __expf(S[r * NT + cc] - mx);
      S[r * NT + cc] = e;
      sum += e;
    }
    const float inv = 1.f / sum;
    for (int cc = 0; cc < NT; ++cc)
      P[r * NT + cc] = (_Float16)(S[r * NT + cc] * inv);
  }
  __syncthreads();

  // O = P[128x128] * V[128x32]: row tile per wave, 2 col tiles, 4 k-steps
#pragma unroll 1
  for (int tj = 0; tj < 2; ++tj) {
    const int ti = wave;
    const int m = ti * 16 + (lane & 15);
    const int n = tj * 16 + (lane & 15);
    v8f acc = {};
#pragma unroll 1
    for (int k0 = 0; k0 < NT; k0 += 32) {
      v16h af = frag_from_lds(&P[m * NT + k0], kbase);   // P row-major in k
      v16h bf = frag_from_lds(&Vt[n * NT + k0], kbase);  // Vt: B[k][n] contiguous in k
      acc = __builtin_amdgcn_wmma_f32_16x16x32_f16(false, af, false, bf,
                                                   (short)0, acc, false, false);
    }
    const int rbase = (lane & 16) ? 8 : 0;
#pragma unroll
    for (int r = 0; r < 8; ++r) {
      const int row = ti * 16 + rbase + r;
      const int col = tj * 16 + (lane & 15);
      const long o = ((long)(b * NT + row) * NV + v) * NHD + h * ND + col;
      Ob[o] = acc[r];
    }
  }
}

// ---------------------------------------------------------------
// LayerNorm over 256 channels, unbiased std, eps on std (torch style).
// Wave32 per row; optionally also emits an f16 copy for the next GEMM.
// ---------------------------------------------------------------
__global__ void __launch_bounds__(256)
layernorm256(const float* __restrict__ in, float* __restrict__ out,
             _Float16* __restrict__ out16,
             const float* __restrict__ g, const float* __restrict__ bta, int rows) {
  const int lane = threadIdx.x & 31;
  const int wave = threadIdx.x >> 5;
  const int row = blockIdx.x * (blockDim.x >> 5) + wave;
  if (row >= rows) return;
  const float* x = in + (long)row * 256;
  float vals[8];
  float s = 0.f, s2 = 0.f;
#pragma unroll
  for (int j = 0; j < 8; ++j) {
    const float t = x[lane + j * 32];
    vals[j] = t; s += t; s2 += t * t;
  }
#pragma unroll
  for (int off = 16; off > 0; off >>= 1) {
    s  += __shfl_xor(s,  off, 32);
    s2 += __shfl_xor(s2, off, 32);
  }
  const float mean = s * (1.f / 256.f);
  const float var  = fmaxf((s2 - 256.f * mean * mean) * (1.f / 255.f), 0.f);
  const float inv  = 1.f / (sqrtf(var) + EPSLN);
  float* o = out + (long)row * 256;
#pragma unroll
  for (int j = 0; j < 8; ++j) {
    const int c = lane + j * 32;
    const float y = g[c] * (vals[j] - mean) * inv + bta[c];
    o[c] = y;
    if (out16) out16[(long)row * 256 + c] = (_Float16)y;
  }
}

// ---------------------------------------------------------------
// ST-GCN contraction: h3[bt,w,c] = sum_{k,v} xw[bt,v,k,c] * A[k,v,w].
// One block per (b,t); xw tile DMA'd to LDS with async b128 copies.
// ---------------------------------------------------------------
__global__ void __launch_bounds__(256)
gcn_contract(const float* __restrict__ xw, const float* __restrict__ A,
             float* __restrict__ h3) {
  extern __shared__ __align__(16) char smem[];
  float* tile = (float*)smem;                  // 25*768 = 19200 f32 (76800 B)
  float* Asm  = tile + NV * NK * NCOUT;        // 1875 f32
  const int bt = blockIdx.x;
  const int tid = threadIdx.x;
  const float* src = xw + (long)bt * NV * NK * NCOUT;
  const uint32_t tile0 = lds_off_u32(tile);
  for (int chunk = tid; chunk < (NV * NK * NCOUT * 4) / 16; chunk += 256)
    async_ld_b128(tile0 + chunk * 16u, (const char*)src + chunk * 16);
  for (int e = tid; e < NK * NV * NV; e += 256) Asm[e] = A[e];
  wait_async0();
  __syncthreads();
  const int c = tid;
#pragma unroll 1
  for (int w = 0; w < NV; ++w) {
    float acc = 0.f;
#pragma unroll
    for (int k = 0; k < NK; ++k)
      for (int v = 0; v < NV; ++v)
        acc += tile[(v * NK + k) * NCOUT + c] * Asm[(k * NV + v) * NV + w];
    h3[((long)bt * NV + w) * NCOUT + c] = acc;
  }
}

// ------------------- small conversion kernels -------------------
__global__ void __launch_bounds__(256)
cvt_f32_f16(const float* __restrict__ s, _Float16* __restrict__ d, long n) {
  const long i = (long)blockIdx.x * blockDim.x + threadIdx.x;
  if (i < n) d[i] = (_Float16)s[i];
}
// W[K,N] f32 -> Wt[N,K] f16 (transposed so B fragments are k-contiguous)
__global__ void __launch_bounds__(256)
cvt_transpose_f16(const float* __restrict__ W, _Float16* __restrict__ Wt, int Kd, int N) {
  const int idx = blockIdx.x * blockDim.x + threadIdx.x;
  if (idx >= Kd * N) return;
  const int k = idx / N, n = idx % N;
  Wt[(long)n * Kd + k] = (_Float16)W[idx];
}

// ---------------------------------------------------------------
extern "C" void kernel_launch(void* const* d_in, const int* in_sizes, int n_in,
                              void* d_out, int out_size, void* d_ws, size_t ws_size,
                              hipStream_t stream) {
  const float* x    = (const float*)d_in[0];
  const float* m    = (const float*)d_in[1];
  const float* Aadj = (const float*)d_in[2];
  const float* Wq1 = (const float*)d_in[4];
  const float* Wk1 = (const float*)d_in[5];
  const float* Wv1 = (const float*)d_in[6];
  const float* g1  = (const float*)d_in[7];
  const float* b1  = (const float*)d_in[8];
  const float* Wq2 = (const float*)d_in[9];
  const float* Wk2 = (const float*)d_in[10];
  const float* Wv2 = (const float*)d_in[11];
  const float* g2  = (const float*)d_in[12];
  const float* b2  = (const float*)d_in[13];
  const float* Wg  = (const float*)d_in[14];
  const float* bg  = (const float*)d_in[15];
  const float* g3  = (const float*)d_in[16];
  const float* b3  = (const float*)d_in[17];
  float* out = (float*)d_out;

  // ---- workspace carving ----
  float* buf0 = (float*)d_ws;                  // q1 -> q2 -> h3
  float* buf1 = buf0 + (long)ROWS * 256;       // k1 -> k2
  float* buf2 = buf1 + (long)ROWS * 256;       // v1 -> v2
  float* buf3 = buf2 + (long)ROWS * 256;       // h1 -> h2 (f32)
  float* buf4 = buf3 + (long)ROWS * 256;       // xw (ROWS x 768)
  _Float16* hb   = (_Float16*)(buf4 + (long)ROWS * 768);
  _Float16* xh   = hb;                          // ROWS*64
  _Float16* mh   = xh + (long)ROWS * 64;        // ROWS*256
  _Float16* actH = mh + (long)ROWS * 256;       // h1/h2 f16 (reused)
  _Float16* wq1t = actH + (long)ROWS * 256;     // [256,64]
  _Float16* wk1t = wq1t + 256 * 64;
  _Float16* wv1t = wk1t + 256 * 64;
  _Float16* wq2t = wv1t + 256 * 64;             // [256,256]
  _Float16* wk2t = wq2t + 256 * 256;
  _Float16* wv2t = wk2t + 256 * 256;
  _Float16* wgt  = wv2t + 256 * 256;            // [768,256]

  const dim3 blk256(256);
  auto gemmGrid = [&](int M, int N) { return dim3((M / 128) * (N / 64)); };
  auto cvtGrid  = [&](long n) { return dim3((unsigned)((n + 255) / 256)); };

  const size_t attnShm = (size_t)3 * NT * ND * 2 + (size_t)NT * NT * 4 + (size_t)NT * NT * 2; // 122880
  const size_t gcnShm  = (size_t)(NV * NK * NCOUT + NK * NV * NV) * sizeof(float);            // 84300

  // ---- precompute f16 operands ----
  cvt_transpose_f16<<<cvtGrid(NCIN * NHD), blk256, 0, stream>>>(Wq1, wq1t, NCIN, NHD);
  cvt_transpose_f16<<<cvtGrid(NCIN * NHD), blk256, 0, stream>>>(Wk1, wk1t, NCIN, NHD);
  cvt_transpose_f16<<<cvtGrid(NCIN * NHD), blk256, 0, stream>>>(Wv1, wv1t, NCIN, NHD);
  cvt_transpose_f16<<<cvtGrid(NHD * NHD),  blk256, 0, stream>>>(Wq2, wq2t, NHD, NHD);
  cvt_transpose_f16<<<cvtGrid(NCM * NHD),  blk256, 0, stream>>>(Wk2, wk2t, NCM, NHD);
  cvt_transpose_f16<<<cvtGrid(NCM * NHD),  blk256, 0, stream>>>(Wv2, wv2t, NCM, NHD);
  cvt_transpose_f16<<<cvtGrid(NHD * NK * NCOUT), blk256, 0, stream>>>(Wg, wgt, NHD, NK * NCOUT);
  cvt_f32_f16<<<cvtGrid((long)ROWS * NCIN), blk256, 0, stream>>>(x, xh, (long)ROWS * NCIN);
  cvt_f32_f16<<<cvtGrid((long)ROWS * NCM),  blk256, 0, stream>>>(m, mh, (long)ROWS * NCM);

  // ---- stage 1: causal self-attention ----
  gemm_wmma_f16<false><<<gemmGrid(ROWS, NHD), blk256, 0, stream>>>(xh, wq1t, nullptr, buf0, ROWS, NHD, NCIN);
  gemm_wmma_f16<false><<<gemmGrid(ROWS, NHD), blk256, 0, stream>>>(xh, wk1t, nullptr, buf1, ROWS, NHD, NCIN);
  gemm_wmma_f16<false><<<gemmGrid(ROWS, NHD), blk256, 0, stream>>>(xh, wv1t, nullptr, buf2, ROWS, NHD, NCIN);
  attn_wmma<<<dim3(NB * NH * NV), blk256, attnShm, stream>>>(buf0, buf1, buf2, buf3, 1);
  layernorm256<<<dim3(ROWS / 8), blk256, 0, stream>>>(buf3, buf3, actH, g1, b1, ROWS);

  // ---- stage 2: cross-attention vs memory ----
  gemm_wmma_f16<false><<<gemmGrid(ROWS, NHD), blk256, 0, stream>>>(actH, wq2t, nullptr, buf0, ROWS, NHD, NHD);
  gemm_wmma_f16<false><<<gemmGrid(ROWS, NHD), blk256, 0, stream>>>(mh,   wk2t, nullptr, buf1, ROWS, NHD, NCM);
  gemm_wmma_f16<false><<<gemmGrid(ROWS, NHD), blk256, 0, stream>>>(mh,   wv2t, nullptr, buf2, ROWS, NHD, NCM);
  attn_wmma<<<dim3(NB * NH * NV), blk256, attnShm, stream>>>(buf0, buf1, buf2, buf3, 0);
  layernorm256<<<dim3(ROWS / 8), blk256, 0, stream>>>(buf3, buf3, actH, g2, b2, ROWS);

  // ---- stage 3: GCN expansion + adjacency contraction ----
  gemm_wmma_f16<true><<<gemmGrid(ROWS, NK * NCOUT), blk256, 0, stream>>>(actH, wgt, bg, buf4, ROWS, NK * NCOUT, NHD);
  gcn_contract<<<dim3(NB * NT), blk256, gcnShm, stream>>>(buf4, Aadj, buf0);
  layernorm256<<<dim3(ROWS / 8), blk256, 0, stream>>>(buf0, out, nullptr, g3, b3, ROWS);
}
